// HGT_63926293234345
// MI455X (gfx1250) — matrix operand order repaired and checked by reading, not compiled, not run on hardware
//
#include <hip/hip_runtime.h>
#include <math.h>

// ---------------------------------------------------------------------------
// HGT forward for MI455X (gfx1250, wave32, WMMA).
// Dense GEMMs run on v_wmma_f32_16x16x32_f16; relation matrices a_rel/m_rel
// are pre-folded into the K/V projection weights so the edge phase is pure
// gather-dot / segment-softmax / scatter-add (bandwidth bound).
// ---------------------------------------------------------------------------

typedef __attribute__((ext_vector_type(16))) _Float16 v16h;
typedef __attribute__((ext_vector_type(8)))  _Float16 v8h;
typedef __attribute__((ext_vector_type(2)))  _Float16 v2h;
typedef __attribute__((ext_vector_type(8)))  float    v8f;

#define HGT_NN   40000        // nodes per type (A == B here)
#define HGT_HID  256
#define HGT_E    300000
#define HGT_SCALE 0.17677669529663687f   // 1/sqrt(32)

// ---------------------------------------------------------------------------
// WMMA GEMM:  out = epilogue(A[MxK] @ W[KxN] + bias)
//   act==1 -> ReLU.  If resid!=null: out = sigmoid(*skip)*x + (1-s)*resid.
// Tiles: BM=64 BN=128 BK=32, 8 waves; wave w -> rows 16*(w>>1), cols 64*(w&1).
// Each wave holds 4 f32 accumulators (16x64 strip) -> 4 v_wmma per K-step.
// B tile is stored transposed in LDS so A and B fragments are contiguous
// 16B reads (ds_load_b128) in the ISA 7.12.2 register layouts:
//   A 16x32 f16 : lane half -> koff 0/8;  af[j] = K{koff..koff+7, 16+koff..}
//   B 32x16 f16 : lane half -> koff 0/8;  bf[j] = K{koff..koff+7, 16+koff..}
// Staging converts fp32->f16 in element pairs (packed ds_store_b32).
// Requires M%64==0, N%128==0, K%32==0 (true for all call sites).
// ---------------------------------------------------------------------------
__global__ __launch_bounds__(256) void hgt_gemm_wmma(
    const float* __restrict__ A, const float* __restrict__ W,
    const float* __restrict__ bias,
    const float* __restrict__ resid, const float* __restrict__ skip_logit,
    float* __restrict__ out, int M, int N, int K, int act)
{
    __shared__ _Float16 As [64][40];    // A tile, row-major   (80B rows, 16B aligned)
    __shared__ _Float16 BsT[128][40];   // B tile, TRANSPOSED: BsT[col][k]

    const int tid  = threadIdx.x;
    const int lane = tid & 31;
    const int wave = tid >> 5;
    const int wrow = wave >> 1;          // 0..3  (16-row group)
    const int wcol = wave & 1;           // 0..1  (64-col group)
    const int row0 = blockIdx.x * 64;
    const int col0 = blockIdx.y * 128;

    v8f acc[4];
    #pragma unroll
    for (int s = 0; s < 4; ++s) acc[s] = (v8f){0.f,0.f,0.f,0.f,0.f,0.f,0.f,0.f};

    const int mrow = wrow * 16 + (lane & 15);
    const int koff = (lane >> 4) * 8;    // A: sub-block offset within 16-K block
    const int kblk = (lane >> 4) * 16;   // unused: B fragment also uses koff
    const int nfr  = lane & 15;

    for (int k0 = 0; k0 < K; k0 += 32) {
        // prefetch next K tile (global_prefetch_b8, near scope)
        if (k0 + 32 < K) {
            __builtin_prefetch(&A[(size_t)(row0 + (tid >> 5) * 8) * K + k0 + 32 + (tid & 31)], 0, 3);
            __builtin_prefetch(&W[(size_t)(k0 + 32 + (tid >> 7) * 16) * N + col0 + (tid & 127)], 0, 3);
        }
        // cooperative fp32 -> f16 staging; packed pair stores (ds_store_b32)
        #pragma unroll
        for (int i = 0; i < 4; ++i) {            // 64x32 A tile, 1024 pairs
            int idx = tid + i * 256;
            int r = idx >> 4, c = (idx & 15) * 2;
            const float* ap = &A[(size_t)(row0 + r) * K + (k0 + c)];
            v2h p = { (_Float16)ap[0], (_Float16)ap[1] };
            *(v2h*)&As[r][c] = p;
        }
        #pragma unroll
        for (int i = 0; i < 8; ++i) {            // 32x128 B tile -> transposed, 2048 pairs
            int idx = tid + i * 256;
            int c = idx & 127, r = (idx >> 7) * 2;
            v2h p = { (_Float16)W[(size_t)(k0 + r)     * N + (col0 + c)],
                      (_Float16)W[(size_t)(k0 + r + 1) * N + (col0 + c)] };
            *(v2h*)&BsT[c][r] = p;
        }
        __syncthreads();

        // A fragment: two aligned 16B LDS loads
        v8h alo = *(const v8h*)&As[mrow][koff];
        v8h ahi = *(const v8h*)&As[mrow][16 + koff];
        v16h af = __builtin_shufflevector(alo, ahi,
                   0,1,2,3,4,5,6,7,8,9,10,11,12,13,14,15);

        #pragma unroll
        for (int s = 0; s < 4; ++s) {
            const int nn = wcol * 64 + s * 16 + nfr;
            v8h blo = *(const v8h*)&BsT[nn][koff];
            v8h bhi = *(const v8h*)&BsT[nn][16 + koff];
            v16h bf = __builtin_shufflevector(blo, bhi,
                       0,1,2,3,4,5,6,7,8,9,10,11,12,13,14,15);
            acc[s] = __builtin_amdgcn_wmma_f32_16x16x32_f16(false, af, false, bf,
                                                            (short)0, acc[s], false, false);
        }
        __syncthreads();
    }

    // epilogue: C layout -> lane n = lane&15, rows (lane>>4)*8 + i
    float beta = 1.0f, ombeta = 0.0f;
    if (skip_logit) {
        float s = __builtin_amdgcn_rcpf(1.0f + __expf(-skip_logit[0]));
        beta = s; ombeta = 1.0f - s;
    }
    const int mbase = (lane >> 4) * 8;
    #pragma unroll
    for (int s = 0; s < 4; ++s) {
        const int col = col0 + wcol * 64 + s * 16 + nfr;
        const float bv = bias[col];
        #pragma unroll
        for (int i = 0; i < 8; ++i) {
            int row = row0 + wrow * 16 + mbase + i;
            float x = acc[s][i] + bv;
            if (act == 1) x = fmaxf(x, 0.0f);
            size_t o = (size_t)row * N + col;
            if (resid) x = beta * x + ombeta * resid[o];
            out[o] = x;
        }
    }
}

// ---------------------------------------------------------------------------
// Fold per-head relation matrix R[8][32][32] into projection weight/bias:
//   W'[k][h*32+f] = sum_d W[k][h*32+d] * R[h][d][f]
// ---------------------------------------------------------------------------
__global__ void hgt_compose_w(const float* __restrict__ W,
                              const float* __restrict__ R,
                              float* __restrict__ Wout)
{
    int idx = blockIdx.x * blockDim.x + threadIdx.x;   // 256*256
    if (idx >= 256 * 256) return;
    int k = idx >> 8, c = idx & 255, h = c >> 5, f = c & 31;
    const float* wr = W + (size_t)k * 256 + h * 32;
    const float* rp = R + h * 1024 + f;
    float s = 0.f;
    #pragma unroll
    for (int d = 0; d < 32; ++d) s += wr[d] * rp[d * 32];
    Wout[idx] = s;
}

__global__ void hgt_compose_b(const float* __restrict__ b,
                              const float* __restrict__ R,
                              float* __restrict__ bout)
{
    int c = threadIdx.x;            // 256
    int h = c >> 5, f = c & 31;
    float s = 0.f;
    #pragma unroll
    for (int d = 0; d < 32; ++d) s += b[h * 32 + d] * R[h * 1024 + d * 32 + f];
    bout[c] = s;
}

// ---------------------------------------------------------------------------
__global__ void hgt_fill(float* __restrict__ p, float v, int n)
{
    int i = blockIdx.x * blockDim.x + threadIdx.x;
    if (i < n) p[i] = v;
}

__global__ void hgt_gelu(float* __restrict__ p, int n)
{
    int i = blockIdx.x * blockDim.x + threadIdx.x;
    if (i < n) {
        float x = p[i];
        p[i] = 0.5f * x * (1.0f + erff(x * 0.70710678118654752f));
    }
}

__device__ inline void atomicMaxF(float* addr, float val)
{
    unsigned int* ua = reinterpret_cast<unsigned int*>(addr);
    unsigned int old = __float_as_uint(*addr);
    while (__uint_as_float(old) < val) {
        unsigned int assumed = old;
        old = atomicCAS(ua, assumed, __float_as_uint(val));
        if (old == assumed) break;
    }
}

// alpha logits: dot(q[dst,h,:], krel[src,h,:]) * p_rel[h] * scale ; seg-max
__global__ void hgt_edge_logits(const int* __restrict__ src, const int* __restrict__ dst,
                                const float* __restrict__ q, const float* __restrict__ krel,
                                const float* __restrict__ prel,
                                float* __restrict__ logits, float* __restrict__ maxb, int E)
{
    int idx = blockIdx.x * blockDim.x + threadIdx.x;
    if (idx >= E * 8) return;
    int e = idx >> 3, h = idx & 7;
    int s = src[e], d = dst[e];
    const float* qp = q    + (size_t)d * HGT_HID + h * 32;
    const float* kp = krel + (size_t)s * HGT_HID + h * 32;
    float acc = 0.f;
    #pragma unroll
    for (int i = 0; i < 32; ++i) acc += qp[i] * kp[i];
    acc *= prel[h] * HGT_SCALE;
    logits[idx] = acc;
    atomicMaxF(&maxb[(size_t)d * 8 + h], acc);
}

// e = exp(l - m); seg-sum
__global__ void hgt_edge_exp(const int* __restrict__ dst, float* __restrict__ logits,
                             const float* __restrict__ maxb, float* __restrict__ sumb, int E)
{
    int idx = blockIdx.x * blockDim.x + threadIdx.x;
    if (idx >= E * 8) return;
    int e = idx >> 3, h = idx & 7;
    int d = dst[e];
    float m = maxb[(size_t)d * 8 + h];
    if (!(m > -1e37f)) m = 0.f;                     // matches isfinite guard
    float w = expf(logits[idx] - m);
    logits[idx] = w;
    atomicAdd(&sumb[(size_t)d * 8 + h], w);
}

// agg[dst] += (e/sum) * vrel[src]
__global__ void hgt_edge_scatter(const int* __restrict__ src, const int* __restrict__ dst,
                                 const float* __restrict__ logits, const float* __restrict__ sumb,
                                 const float* __restrict__ vrel, float* __restrict__ agg, int E)
{
    int idx = blockIdx.x * blockDim.x + threadIdx.x;
    if (idx >= E * 8) return;
    int e = idx >> 3, h = idx & 7;
    int s = src[e], d = dst[e];
    float coef = logits[idx] / (sumb[(size_t)d * 8 + h] + 1e-16f);
    const float* vp = vrel + (size_t)s * HGT_HID + h * 32;
    float*       ap = agg  + (size_t)d * HGT_HID + h * 32;
    #pragma unroll
    for (int i = 0; i < 32; ++i) atomicAdd(&ap[i], coef * vp[i]);
}

// ---------------------------------------------------------------------------
extern "C" void kernel_launch(void* const* d_in, const int* in_sizes, int n_in,
                              void* d_out, int out_size, void* d_ws, size_t ws_size,
                              hipStream_t stream)
{
    (void)in_sizes; (void)n_in; (void)out_size; (void)ws_size;

    const float* xA  = (const float*)d_in[0];
    const float* xB  = (const float*)d_in[1];
    const int*   eAB = (const int*)d_in[2];    // [2,E]: row0=src(A), row1=dst(B)
    const int*   eBA = (const int*)d_in[3];

    auto P = [&](int i) { return (const float*)d_in[i]; };

    // params tree_leaves order (sorted dict keys):
    //   layers[0..1] each 24 leaves, then lin_in (4 leaves)
    const int PB = 4, PL = 24;
    const int LIN = PB + 2 * PL;               // 52
    // per-layer leaf offsets
    enum {
        A_LIN_A_B = 0, A_LIN_A_W, A_LIN_B_B, A_LIN_B_W,
        AREL_AB, AREL_BA,
        K_A_B, K_A_W, Q_A_B, Q_A_W, V_A_B, V_A_W,
        K_B_B, K_B_W, Q_B_B, Q_B_W, V_B_B, V_B_W,
        MREL_AB, MREL_BA, PREL_AB, PREL_BA, SKIP_A, SKIP_B
    };

    const size_t NH = (size_t)HGT_NN * HGT_HID;    // 10.24M floats
    float* ws     = (float*)d_ws;
    float* hA     = ws;
    float* hB     = hA + NH;
    float* qbuf   = hB + NH;
    float* kbuf   = qbuf + NH;
    float* vbuf   = kbuf + NH;
    float* aggA   = vbuf + NH;
    float* aggB   = aggA + NH;
    float* logits = aggB + NH;                     // E*8
    float* maxb   = logits + (size_t)HGT_E * 8;    // N*8
    float* sumb   = maxb + (size_t)HGT_NN * 8;
    float* wcomp  = sumb + (size_t)HGT_NN * 8;     // 256*256
    float* bcomp  = wcomp + 256 * 256;             // 256

    const dim3 blk(256);
    auto gemm = [&](const float* Ain, const float* Wp, const float* bp,
                    const float* resid, const float* skip, float* op,
                    int M, int N, int K, int act) {
        dim3 grid(M / 64, N / 128);
        hgt_gemm_wmma<<<grid, blk, 0, stream>>>(Ain, Wp, bp, resid, skip, op, M, N, K, act);
    };
    auto fill = [&](float* p, float v, int n) {
        hgt_fill<<<dim3((n + 255) / 256), blk, 0, stream>>>(p, v, n);
    };
    const int EG = (HGT_E * 8 + 255) / 256;
    auto edge_pass = [&](const int* eidx, const float* q, const float* kr,
                         const float* vr, const float* prel, float* agg) {
        const int* src = eidx;
        const int* dst = eidx + HGT_E;
        fill(maxb, -INFINITY, HGT_NN * 8);
        fill(sumb, 0.0f, HGT_NN * 8);
        hgt_edge_logits <<<dim3(EG), blk, 0, stream>>>(src, dst, q, kr, prel, logits, maxb, HGT_E);
        hgt_edge_exp    <<<dim3(EG), blk, 0, stream>>>(dst, logits, maxb, sumb, HGT_E);
        hgt_edge_scatter<<<dim3(EG), blk, 0, stream>>>(src, dst, logits, sumb, vr, agg, HGT_E);
    };

    // ---- input projections: h = relu(x @ W + b)
    gemm(xA, P(LIN + 1), P(LIN + 0), nullptr, nullptr, hA, HGT_NN, 256, 128, 1);
    gemm(xB, P(LIN + 3), P(LIN + 2), nullptr, nullptr, hB, HGT_NN, 256, 64, 1);

    for (int L = 0; L < 2; ++L) {
        const int b0 = PB + L * PL;
        fill(aggA, 0.0f, (int)NH);
        fill(aggB, 0.0f, (int)NH);

        // ---- edge type AB: src=A, dst=B ----
        gemm(hB, P(b0 + Q_B_W), P(b0 + Q_B_B), nullptr, nullptr, qbuf, HGT_NN, 256, 256, 0);
        hgt_compose_w<<<dim3(256), blk, 0, stream>>>(P(b0 + K_A_W), P(b0 + AREL_AB), wcomp);
        hgt_compose_b<<<dim3(1),   blk, 0, stream>>>(P(b0 + K_A_B), P(b0 + AREL_AB), bcomp);
        gemm(hA, wcomp, bcomp, nullptr, nullptr, kbuf, HGT_NN, 256, 256, 0);
        hgt_compose_w<<<dim3(256), blk, 0, stream>>>(P(b0 + V_A_W), P(b0 + MREL_AB), wcomp);
        hgt_compose_b<<<dim3(1),   blk, 0, stream>>>(P(b0 + V_A_B), P(b0 + MREL_AB), bcomp);
        gemm(hA, wcomp, bcomp, nullptr, nullptr, vbuf, HGT_NN, 256, 256, 0);
        edge_pass(eAB, qbuf, kbuf, vbuf, P(b0 + PREL_AB), aggB);

        // ---- edge type BA: src=B, dst=A ----
        gemm(hA, P(b0 + Q_A_W), P(b0 + Q_A_B), nullptr, nullptr, qbuf, HGT_NN, 256, 256, 0);
        hgt_compose_w<<<dim3(256), blk, 0, stream>>>(P(b0 + K_B_W), P(b0 + AREL_BA), wcomp);
        hgt_compose_b<<<dim3(1),   blk, 0, stream>>>(P(b0 + K_B_B), P(b0 + AREL_BA), bcomp);
        gemm(hB, wcomp, bcomp, nullptr, nullptr, kbuf, HGT_NN, 256, 256, 0);
        hgt_compose_w<<<dim3(256), blk, 0, stream>>>(P(b0 + V_B_W), P(b0 + MREL_BA), wcomp);
        hgt_compose_b<<<dim3(1),   blk, 0, stream>>>(P(b0 + V_B_B), P(b0 + MREL_BA), bcomp);
        gemm(hB, wcomp, bcomp, nullptr, nullptr, vbuf, HGT_NN, 256, 256, 0);
        edge_pass(eBA, qbuf, kbuf, vbuf, P(b0 + PREL_BA), aggA);

        // ---- output: h = beta*(gelu(agg)@Wa + ba) + (1-beta)*h  (in place)
        hgt_gelu<<<dim3((int)(NH / 256)), blk, 0, stream>>>(aggA, (int)NH);
        hgt_gelu<<<dim3((int)(NH / 256)), blk, 0, stream>>>(aggB, (int)NH);
        gemm(aggA, P(b0 + A_LIN_A_W), P(b0 + A_LIN_A_B), hA, P(b0 + SKIP_A), hA, HGT_NN, 256, 256, 0);
        gemm(aggB, P(b0 + A_LIN_B_W), P(b0 + A_LIN_B_B), hB, P(b0 + SKIP_B), hB, HGT_NN, 256, 256, 0);
    }

    hipMemcpyAsync(d_out, hA, NH * sizeof(float), hipMemcpyDeviceToDevice, stream);
    hipMemcpyAsync((float*)d_out + NH, hB, NH * sizeof(float), hipMemcpyDeviceToDevice, stream);
}
